// DenoisingNet_MLP_3_54623394070973
// MI455X (gfx1250) — compile-verified, matrix operand order checked by reading, not run on hardware
//
#include <hip/hip_runtime.h>
#include <math.h>

// ---------------------------------------------------------------------------
// Types for CDNA5 WMMA
// ---------------------------------------------------------------------------
typedef __bf16 bf16_t;
typedef __attribute__((ext_vector_type(16))) __bf16 v16bf;
typedef __attribute__((ext_vector_type(8)))  float  v8f;

#define CDIV(a, b) (((a) + (b) - 1) / (b))

// ---------------------------------------------------------------------------
// Problem constants
// ---------------------------------------------------------------------------
static const long long LP  = 113LL * 113LL;  // 12769 stride-1 patches per image
static const long long NBt = 2;              // batch
static const int DDIM = 624;                 // 512 + 112

// Workspace layout (float offsets). tmp1 aliases zA/zB (disjoint live ranges).
static const long long UF_OFF    = 0;                                   // [2*L,256]
static const long long THR_OFF   = UF_OFF    + NBt * LP * 256;          // [2,L,624]
static const long long Y_OFF     = THR_OFF   + NBt * LP * 624;          // [2,L,624]
static const long long ZA_OFF    = Y_OFF     + NBt * LP * 624;          // [2,L,624]
static const long long ZB_OFF    = ZA_OFF    + NBt * LP * 624;          // [2,L,624]
static const long long TMP1_OFF  = ZA_OFF;                              // [2*L,1024] alias
static const long long TMP2_OFF  = ZB_OFF    + NBt * LP * 624;          // [2*L,512]
static const long long XP_OFF    = TMP2_OFF;                            // [2,L,256] alias
static const long long WG_OFF    = TMP2_OFF  + NBt * LP * 512;          // [2,L,256]
static const long long S_OFF     = WG_OFF    + NBt * LP * 256;          // [2,624,624]
static const long long DCAT_OFF  = S_OFF     + NBt * 624 * 624;         // [2,256,624]
static const long long EXT_OFF   = DCAT_OFF  + NBt * 256 * 624;         // [224,256]
static const long long SD_OFF    = EXT_OFF   + 224 * 256;               // [224,256]
static const long long SDT1_OFF  = SD_OFF    + 224 * 256;               // [224,1024]
static const long long SDT2_OFF  = SDT1_OFF  + 224 * 1024;              // [224,1024]
static const long long POOL_OFF  = SDT2_OFF  + 224 * 1024;              // [4,256]
static const long long CAH_OFF   = POOL_OFF  + 4 * 256;                 // [4,512]
static const long long CAO_OFF   = CAH_OFF   + 4 * 512;                 // [4,256]
static const long long CA_OFF    = CAO_OFF   + 4 * 256;                 // [2,256]
static const long long SPOOL_OFF = CA_OFF    + 2 * 256;                 // [2,2,112]
static const long long SA_OFF    = SPOOL_OFF + 2 * 2 * 112;             // [2,112]
static const long long RES_OFF   = SA_OFF    + 2 * 112;                 // [2,128,128]

// ---------------------------------------------------------------------------
// CDNA5 async global->LDS copy (ASYNCcnt path), inline asm per ISA ch.10/15.18
// ---------------------------------------------------------------------------
__device__ __forceinline__ void async_ld_b128(unsigned lds_off, const void* gaddr) {
  asm volatile("global_load_async_to_lds_b128 %0, %1, off"
               :: "v"(lds_off), "v"(gaddr) : "memory");
}
__device__ __forceinline__ void wait_async0() {
  asm volatile("s_wait_asynccnt 0x0" ::: "memory");
}

// ---------------------------------------------------------------------------
// Generic batched WMMA GEMM:  C = epilogue(A @ B)
//   block tile 128x64, 8 wave32 waves, each wave 32x32 via 4x v_wmma 16x16x32.
//   fp32 tiles staged in LDS (async-to-LDS on interior tiles), converted to
//   bf16 at fragment build.  Padded stride 36 floats (144 B = 9*16 B).
// ---------------------------------------------------------------------------
struct GemmP {
  const float* A; const float* B; float* C;
  const float* bias;       // per-output-column, may be null
  const float* E1;         // extra matrix 1 (y for LISTA, Wg for clipmul)
  const float* E2;         // extra matrix 2 (thr for LISTA)
  const float* cPtr;       // device scalar c (for 1/c scaling), may be null
  long long sA, sB, sC, sE1, sE2;   // per-batch element strides
  int lda, ldb, ldc, ldE1, ldE2;
  int M, N, K;
  float alphaMul; int alphaInvC;    // alpha = alphaMul * (alphaInvC ? 1/c : 1)
  int act;                 // 0 none, 1 relu, 2 sigmoid   (EP==0 only)
  int addI;                // add identity (m==n)         (EP==0 only)
};

__device__ __forceinline__ v16bf frag_from_lds(const float* rowbase, int half) {
  // element e<8 -> k = 8*half + e ; e>=8 -> k = 16 + 8*half + (e-8)
  float4 f0 = *(const float4*)(rowbase + half * 8 + 0);
  float4 f1 = *(const float4*)(rowbase + half * 8 + 4);
  float4 f2 = *(const float4*)(rowbase + half * 8 + 16);
  float4 f3 = *(const float4*)(rowbase + half * 8 + 20);
  v16bf a;
  a[0]  = (bf16_t)f0.x; a[1]  = (bf16_t)f0.y; a[2]  = (bf16_t)f0.z; a[3]  = (bf16_t)f0.w;
  a[4]  = (bf16_t)f1.x; a[5]  = (bf16_t)f1.y; a[6]  = (bf16_t)f1.z; a[7]  = (bf16_t)f1.w;
  a[8]  = (bf16_t)f2.x; a[9]  = (bf16_t)f2.y; a[10] = (bf16_t)f2.z; a[11] = (bf16_t)f2.w;
  a[12] = (bf16_t)f3.x; a[13] = (bf16_t)f3.y; a[14] = (bf16_t)f3.z; a[15] = (bf16_t)f3.w;
  return a;
}

template <int TRANSA, int TRANSB, int EP>
__global__ __launch_bounds__(256) void k_gemm(GemmP p) {
  __shared__ __align__(16) float Asf[128 * 36];  // [m][k], stride 36 floats
  __shared__ __align__(16) float Bsf[64 * 36];   // [n][k] (B transposed)

  const int tid   = threadIdx.x;
  const int lane  = tid & 31;
  const int wave  = tid >> 5;
  const int waveM = wave >> 1;      // 0..3
  const int waveN = wave & 1;       // 0..1
  const int half  = lane >> 4;      // 0/1
  const int l16   = lane & 15;
  const int m0 = blockIdx.y * 128;
  const int n0 = blockIdx.x * 64;
  const int b  = blockIdx.z;

  const float* A = p.A + (long long)b * p.sA;
  const float* B = p.B + (long long)b * p.sB;
  float*       C = p.C + (long long)b * p.sC;

  v8f acc[2][2];
#pragma unroll
  for (int i = 0; i < 2; ++i)
#pragma unroll
    for (int j = 0; j < 2; ++j)
#pragma unroll
      for (int v = 0; v < 8; ++v) acc[i][j][v] = 0.0f;

  for (int kb = 0; kb < p.K; kb += 32) {
    __syncthreads();
    const bool fullK = (kb + 32 <= p.K);

    // ---- stage A tile -> Asf[m][k] ----
    if (TRANSA == 0) {
      if (fullK && (m0 + 128 <= p.M)) {
        // interior: CDNA5 async global->LDS, no VGPR round trip
#pragma unroll
        for (int it = 0; it < 4; ++it) {
          int slot = tid + 256 * it;
          int ml = slot >> 3, kq = (slot & 7) << 2;
          async_ld_b128((unsigned)(uintptr_t)&Asf[ml * 36 + kq],
                        A + (long long)(m0 + ml) * p.lda + (kb + kq));
        }
      } else {
#pragma unroll
        for (int it = 0; it < 4; ++it) {
          int slot = tid + 256 * it;
          int ml = slot >> 3, kq = (slot & 7) << 2;
          int gm = m0 + ml, gk = kb + kq;
          float* dst = &Asf[ml * 36 + kq];
          if (gm < p.M && gk + 3 < p.K) {
            *(float4*)dst = *(const float4*)(A + (long long)gm * p.lda + gk);
          } else {
#pragma unroll
            for (int j = 0; j < 4; ++j)
              dst[j] = (gm < p.M && gk + j < p.K)
                         ? A[(long long)gm * p.lda + gk + j] : 0.0f;
          }
        }
      }
    } else {  // A stored [K, M]: synchronous transpose staging
#pragma unroll
      for (int it = 0; it < 4; ++it) {
        int slot = tid + 256 * it;
        int kl = slot >> 5, mq = (slot & 31) << 2;
        int gk = kb + kl;
        if (gk < p.K && m0 + mq + 3 < p.M) {
          float4 v = *(const float4*)(A + (long long)gk * p.lda + m0 + mq);
          Asf[(mq + 0) * 36 + kl] = v.x;
          Asf[(mq + 1) * 36 + kl] = v.y;
          Asf[(mq + 2) * 36 + kl] = v.z;
          Asf[(mq + 3) * 36 + kl] = v.w;
        } else {
#pragma unroll
          for (int j = 0; j < 4; ++j) {
            int gm = m0 + mq + j;
            Asf[(mq + j) * 36 + kl] = (gk < p.K && gm < p.M)
                                        ? A[(long long)gk * p.lda + gm] : 0.0f;
          }
        }
      }
    }

    // ---- stage B tile -> Bsf[n][k] ----
    if (TRANSB == 0) {  // B stored [K, N]: synchronous transpose staging
#pragma unroll
      for (int it = 0; it < 2; ++it) {
        int slot = tid + 256 * it;
        int kl = slot >> 4, nq = (slot & 15) << 2;
        int gk = kb + kl;
        if (gk < p.K && n0 + nq + 3 < p.N) {
          float4 v = *(const float4*)(B + (long long)gk * p.ldb + n0 + nq);
          Bsf[(nq + 0) * 36 + kl] = v.x;
          Bsf[(nq + 1) * 36 + kl] = v.y;
          Bsf[(nq + 2) * 36 + kl] = v.z;
          Bsf[(nq + 3) * 36 + kl] = v.w;
        } else {
#pragma unroll
          for (int j = 0; j < 4; ++j) {
            int gn = n0 + nq + j;
            Bsf[(nq + j) * 36 + kl] = (gk < p.K && gn < p.N)
                                        ? B[(long long)gk * p.ldb + gn] : 0.0f;
          }
        }
      }
    } else {  // B stored [N, K]: k-contiguous, async on interior
      if (fullK && (n0 + 64 <= p.N)) {
#pragma unroll
        for (int it = 0; it < 2; ++it) {
          int slot = tid + 256 * it;
          int nl = slot >> 3, kq = (slot & 7) << 2;
          async_ld_b128((unsigned)(uintptr_t)&Bsf[nl * 36 + kq],
                        B + (long long)(n0 + nl) * p.ldb + (kb + kq));
        }
      } else {
#pragma unroll
        for (int it = 0; it < 2; ++it) {
          int slot = tid + 256 * it;
          int nl = slot >> 3, kq = (slot & 7) << 2;
          int gn = n0 + nl, gk = kb + kq;
          float* dst = &Bsf[nl * 36 + kq];
          if (gn < p.N && gk + 3 < p.K) {
            *(float4*)dst = *(const float4*)(B + (long long)gn * p.ldb + gk);
          } else {
#pragma unroll
            for (int j = 0; j < 4; ++j)
              dst[j] = (gn < p.N && gk + j < p.K)
                         ? B[(long long)gn * p.ldb + gk + j] : 0.0f;
          }
        }
      }
    }

    wait_async0();       // ASYNCcnt == 0: all async tile data landed in LDS
    __syncthreads();

    // ---- fragments (fp32 LDS -> bf16) + 4x WMMA ----
    v16bf afr[2], bfr[2];
#pragma unroll
    for (int mt = 0; mt < 2; ++mt)
      afr[mt] = frag_from_lds(&Asf[(waveM * 32 + mt * 16 + l16) * 36], half);
#pragma unroll
    for (int nt = 0; nt < 2; ++nt)
      bfr[nt] = frag_from_lds(&Bsf[(waveN * 32 + nt * 16 + l16) * 36], half);
#pragma unroll
    for (int mt = 0; mt < 2; ++mt)
#pragma unroll
      for (int nt = 0; nt < 2; ++nt)
        acc[mt][nt] = __builtin_amdgcn_wmma_f32_16x16x32_bf16(
            false, afr[mt], false, bfr[nt], (short)0, acc[mt][nt], false, false);
  }

  // ---- epilogue ----
  float alpha = p.alphaMul;
  if (p.alphaInvC) alpha *= 1.0f / *p.cPtr;
  float invc = (EP == 1) ? (1.0f / *p.cPtr) : 1.0f;
  const float* E1 = p.E1 ? p.E1 + (long long)b * p.sE1 : nullptr;
  const float* E2 = p.E2 ? p.E2 + (long long)b * p.sE2 : nullptr;

#pragma unroll
  for (int mt = 0; mt < 2; ++mt)
#pragma unroll
    for (int nt = 0; nt < 2; ++nt) {
      int gn = n0 + waveN * 32 + nt * 16 + l16;
      if (gn >= p.N) continue;
      float bv = p.bias ? p.bias[gn] : 0.0f;
#pragma unroll
      for (int v = 0; v < 8; ++v) {
        int gm = m0 + waveM * 32 + mt * 16 + v + 8 * half;
        if (gm >= p.M) continue;
        float r = acc[mt][nt][v];
        if (EP == 0) {
          r = (r + bv) * alpha;
          if (p.addI && gm == gn) r += 1.0f;
          if (p.act == 1) r = fmaxf(r, 0.0f);
          else if (p.act == 2) r = 1.0f / (1.0f + __expf(-r));
        } else if (EP == 1) {  // LISTA: soft(acc + y/c, thr)
          float x = r + E1[(long long)gm * p.ldE1 + gn] * invc;
          float l = E2[(long long)gm * p.ldE2 + gn];
          float a = fabsf(x) - l; a = a > 0.0f ? a : 0.0f;
          r = x > 0.0f ? a : (x < 0.0f ? -a : 0.0f);
        } else {               // clip01 * Wg
          float x = fminf(fmaxf(r, 0.0f), 1.0f);
          r = x * E1[(long long)gm * p.ldE1 + gn];
        }
        C[(long long)gm * p.ldc + gn] = r;
      }
    }
}

// ---------------------------------------------------------------------------
// Small helper kernels
// ---------------------------------------------------------------------------
__global__ void k_unfold(const float* __restrict__ img, float* __restrict__ uf) {
  long long i = (long long)blockIdx.x * 256 + threadIdx.x;
  if (i >= NBt * LP * 256) return;
  int q = (int)(i & 255);
  long long t = i >> 8;
  int pidx = (int)(t % LP);
  int n    = (int)(t / LP);
  int pi = pidx / 113, pj = pidx % 113;
  int r = q >> 4, c2 = q & 15;
  uf[i] = img[(long long)n * 16384 + (pi + r) * 128 + (pj + c2)];
}

__global__ void k_ext(const float* __restrict__ img, float* __restrict__ ext) {
  long long i = (long long)blockIdx.x * 256 + threadIdx.x;
  if (i >= NBt * 112 * 256) return;
  int q = (int)(i & 255);
  long long t = i >> 8;
  int s = (int)(t % 112);
  int n = (int)(t / 112);
  int orig = 2 * s;                 // step=2 subsample of 15x15 stride-8 grid
  int i8 = orig / 15, j8 = orig % 15;
  int r = q >> 4, c2 = q & 15;
  ext[i] = img[(long long)n * 16384 + (8 * i8 + r) * 128 + (8 * j8 + c2)];
}

__global__ void k_rownorm(float* x) {  // 224 rows of 256
  __shared__ float red[256];
  int row = blockIdx.x, tid = threadIdx.x;
  float v = x[(long long)row * 256 + tid];
  red[tid] = v * v; __syncthreads();
  for (int s = 128; s > 0; s >>= 1) {
    if (tid < s) red[tid] += red[tid + s];
    __syncthreads();
  }
  float sc = 1.0f / fmaxf(sqrtf(red[0]), 1e-12f);
  x[(long long)row * 256 + tid] = v * sc;
}

__global__ void k_chpool(const float* __restrict__ sd, float* __restrict__ pool) {
  int id = blockIdx.x * 256 + threadIdx.x;
  if (id >= 512) return;
  int n = id >> 8, ch = id & 255;
  float s = 0.0f, m = -3.4e38f;
  for (int t = 0; t < 112; ++t) {
    float v = sd[((long long)(n * 112 + t)) * 256 + ch];
    s += v; m = fmaxf(m, v);
  }
  pool[(n * 2 + 0) * 256 + ch] = s / 112.0f;
  pool[(n * 2 + 1) * 256 + ch] = m;
}

__global__ void k_cacomb(const float* __restrict__ cao, float* __restrict__ ca) {
  int id = blockIdx.x * 256 + threadIdx.x;
  if (id >= 512) return;
  int n = id >> 8, ch = id & 255;
  float v = cao[(n * 2 + 0) * 256 + ch] + cao[(n * 2 + 1) * 256 + ch];
  ca[id] = 1.0f / (1.0f + __expf(-v));
}

__global__ void k_spool(const float* __restrict__ sd, const float* __restrict__ ca,
                        float* __restrict__ spool) {  // grid = 224 = n*112+s
  __shared__ float ssum[256];
  __shared__ float smx[256];
  int b = blockIdx.x, tid = threadIdx.x;
  int n = b / 112, s = b % 112;
  float v = sd[(long long)b * 256 + tid] * ca[n * 256 + tid];
  ssum[tid] = v; smx[tid] = v; __syncthreads();
  for (int t = 128; t > 0; t >>= 1) {
    if (tid < t) {
      ssum[tid] += ssum[tid + t];
      smx[tid] = fmaxf(smx[tid], smx[tid + t]);
    }
    __syncthreads();
  }
  if (tid == 0) {
    spool[(n * 2 + 0) * 112 + s] = ssum[0] / 256.0f;
    spool[(n * 2 + 1) * 112 + s] = smx[0];
  }
}

__global__ void k_saconv(const float* __restrict__ spool,
                         const float* __restrict__ saw, float* __restrict__ sa) {
  int id = threadIdx.x;
  if (id >= 224) return;
  int n = id / 112, s = id % 112;
  int y = s / 14, x = s % 14;
  float acc = 0.0f;
  for (int ic = 0; ic < 2; ++ic)
    for (int ky = 0; ky < 7; ++ky) {
      int yy = y + ky - 3;
      if (yy < 0 || yy >= 8) continue;
      for (int kx = 0; kx < 7; ++kx) {
        int xx = x + kx - 3;
        if (xx < 0 || xx >= 14) continue;
        acc += spool[(n * 2 + ic) * 112 + yy * 14 + xx] * saw[ic * 49 + ky * 7 + kx];
      }
    }
  sa[id] = 1.0f / (1.0f + __expf(-acc));
}

__global__ void k_dcat(const float* __restrict__ Dict, const float* __restrict__ sd,
                       const float* __restrict__ ca, const float* __restrict__ sa,
                       float* __restrict__ dcat) {
  long long i = (long long)blockIdx.x * 256 + threadIdx.x;
  if (i >= NBt * 256 * DDIM) return;
  int col = (int)(i % DDIM);
  int ch  = (int)((i / DDIM) % 256);
  int n   = (int)(i / ((long long)DDIM * 256));
  float v;
  if (col < 512) {
    v = Dict[(long long)ch * 512 + col];
  } else {
    int s = col - 512;
    v = sd[((long long)(n * 112 + s)) * 256 + ch] * ca[n * 256 + ch] * sa[n * 112 + s];
  }
  dcat[i] = v;
}

__global__ void k_soft0(const float* __restrict__ y, const float* __restrict__ thr,
                        float* __restrict__ z, long long total) {
  long long i = (long long)blockIdx.x * 256 + threadIdx.x;
  if (i >= total) return;
  float x = y[i], l = thr[i];
  float a = fabsf(x) - l; a = a > 0.0f ? a : 0.0f;
  z[i] = x > 0.0f ? a : (x < 0.0f ? -a : 0.0f);
}

__global__ void k_fold_div(const float* __restrict__ xp, const float* __restrict__ wg,
                           float* __restrict__ out) {
  int i = blockIdx.x * 256 + threadIdx.x;
  if (i >= 2 * 16384) return;
  int n = i >> 14, hw = i & 16383;
  int h = hw >> 7, w = hw & 127;
  float num = 0.0f, den = 0.0f;
  int r0 = h - 112 > 0 ? h - 112 : 0, r1 = h < 15 ? h : 15;
  for (int r = r0; r <= r1; ++r) {
    int pi = h - r;
    int c0 = w - 112 > 0 ? w - 112 : 0, c1 = w < 15 ? w : 15;
    for (int c = c0; c <= c1; ++c) {
      int pj = w - c;
      long long pq = (((long long)n * LP + (long long)pi * 113 + pj) << 8) + (r << 4) + c;
      num += xp[pq]; den += wg[pq];
    }
  }
  out[i] = num / den;
}

// ---------------------------------------------------------------------------
// Host-side helpers
// ---------------------------------------------------------------------------
static GemmP mkgemm(const float* A, int lda, long long sA,
                    const float* B, int ldb, long long sB,
                    float* C, int ldc, long long sC, int M, int N, int K) {
  GemmP p{};
  p.A = A; p.lda = lda; p.sA = sA;
  p.B = B; p.ldb = ldb; p.sB = sB;
  p.C = C; p.ldc = ldc; p.sC = sC;
  p.M = M; p.N = N; p.K = K;
  p.alphaMul = 1.0f;
  return p;
}

template <int TA, int TB, int EP>
static void launch_gemm_t(hipStream_t st, const GemmP& p, int batch) {
  dim3 g(CDIV(p.N, 64), CDIV(p.M, 128), batch);
  hipLaunchKernelGGL((k_gemm<TA, TB, EP>), g, dim3(256), 0, st, p);
}

static void mlp_layer(hipStream_t st, const float* A, int M, int K, int N,
                      const float* W, const float* b, float* C, int act) {
  GemmP p = mkgemm(A, K, 0, W, N, 0, C, N, 0, M, N, K);
  p.bias = b; p.act = act;
  launch_gemm_t<0, 0, 0>(st, p, 1);
}

struct StageW {
  const float *s1w,*s1b,*s2w,*s2b,*s3w,*s3b,*s4w,*s4b;
  const float *caw1,*caw2,*saw;
  const float *l1w,*l1b,*l2w,*l2b,*l3w,*l3b;   // lam chain -> thr cols 512..623
  const float *p1w,*p1b,*p2w,*p2b,*p3w,*p3b;   // pd  chain -> thr cols 0..511
  const float *w1w,*w1b,*w2w,*w2b,*w3w,*w3b;
  const float *Dict, *cPtr;
};

static void run_stage(hipStream_t st, const float* img, float* outimg,
                      float* ws, const StageW& W) {
  float* UF   = ws + UF_OFF;
  float* THR  = ws + THR_OFF;
  float* Y    = ws + Y_OFF;
  float* ZA   = ws + ZA_OFF;
  float* ZB   = ws + ZB_OFF;
  float* TMP1 = ws + TMP1_OFF;
  float* TMP2 = ws + TMP2_OFF;
  float* XP   = ws + XP_OFF;
  float* WG   = ws + WG_OFF;
  float* Smat = ws + S_OFF;
  float* DCAT = ws + DCAT_OFF;
  float* EXT  = ws + EXT_OFF;
  float* SD   = ws + SD_OFF;
  float* SDT1 = ws + SDT1_OFF;
  float* SDT2 = ws + SDT2_OFF;
  float* POOL = ws + POOL_OFF;
  float* CAH  = ws + CAH_OFF;
  float* CAO  = ws + CAO_OFF;
  float* CA   = ws + CA_OFF;
  float* SPOOL= ws + SPOOL_OFF;
  float* SA   = ws + SA_OFF;

  const int ML = (int)(NBt * LP);  // 25538 rows for weight-shared MLPs

  // unfold (stride 1) + subsampled stride-8 patches
  k_unfold<<<(int)CDIV(NBt * LP * 256, 256), 256, 0, st>>>(img, UF);
  k_ext<<<(int)CDIV(NBt * 112 * 256, 256), 256, 0, st>>>(img, EXT);

  // sdict MLP: 256->512->1024->512->256, relu between, none at end
  mlp_layer(st, EXT,  224, 256,  512,  W.s1w, W.s1b, SDT1, 1);
  mlp_layer(st, SDT1, 224, 512,  1024, W.s2w, W.s2b, SDT2, 1);
  mlp_layer(st, SDT2, 224, 1024, 512,  W.s3w, W.s3b, SDT1, 1);
  mlp_layer(st, SDT1, 224, 512,  256,  W.s4w, W.s4b, SD,   0);
  k_rownorm<<<224, 256, 0, st>>>(SD);

  // CBAM channel attention (shared MLP on avg+max pool, no bias)
  k_chpool<<<2, 256, 0, st>>>(SD, POOL);
  mlp_layer(st, POOL, 4, 256, 512, W.caw1, nullptr, CAH, 1);
  mlp_layer(st, CAH,  4, 512, 256, W.caw2, nullptr, CAO, 0);
  k_cacomb<<<2, 256, 0, st>>>(CAO, CA);
  // spatial attention: pools over channels + 7x7 conv + sigmoid
  k_spool<<<224, 256, 0, st>>>(SD, CA, SPOOL);
  k_saconv<<<1, 256, 0, st>>>(SPOOL, W.saw, SA);
  // Dcat = [Dict | sd*ca*sa]
  k_dcat<<<(int)CDIV(NBt * 256 * DDIM, 256), 256, 0, st>>>(W.Dict, SD, CA, SA, DCAT);

  // thresholds: pd chain -> cols 0..511 (scaled by 1/c)
  mlp_layer(st, UF,   ML, 256,  1024, W.p1w, W.p1b, TMP1, 1);
  mlp_layer(st, TMP1, ML, 1024, 512,  W.p2w, W.p2b, TMP2, 1);
  {
    GemmP p = mkgemm(TMP2, 512, 0, W.p3w, 512, 0, THR, DDIM, 0, ML, 512, 512);
    p.bias = W.p3b; p.alphaInvC = 1; p.cPtr = W.cPtr;
    launch_gemm_t<0, 0, 0>(st, p, 1);
  }
  // lam chain -> cols 512..623 (scaled by 1/c)
  mlp_layer(st, UF,   ML, 256,  1024, W.l1w, W.l1b, TMP1, 1);
  mlp_layer(st, TMP1, ML, 1024, 512,  W.l2w, W.l2b, TMP2, 1);
  {
    GemmP p = mkgemm(TMP2, 512, 0, W.l3w, 112, 0, THR + 512, DDIM, 0, ML, 112, 512);
    p.bias = W.l3b; p.alphaInvC = 1; p.cPtr = W.cPtr;
    launch_gemm_t<0, 0, 0>(st, p, 1);
  }
  // gating Wg = sigmoid(MLP(uf))
  mlp_layer(st, UF,   ML, 256,  1024, W.w1w, W.w1b, TMP1, 1);
  mlp_layer(st, TMP1, ML, 1024, 512,  W.w2w, W.w2b, TMP2, 1);
  mlp_layer(st, TMP2, ML, 512,  256,  W.w3w, W.w3b, WG,   2);

  // y = uf @ Dcat  (batched over N=2: Dcat differs per image)
  {
    GemmP p = mkgemm(UF, 256, LP * 256, DCAT, DDIM, 256LL * DDIM,
                     Y, DDIM, LP * DDIM, (int)LP, DDIM, 256);
    launch_gemm_t<0, 0, 0>(st, p, 2);
  }
  // S = I - Dcat^T Dcat / c  (symmetric); A = Dcat stored [K=256, M=624]
  {
    GemmP p = mkgemm(DCAT, DDIM, 256LL * DDIM, DCAT, DDIM, 256LL * DDIM,
                     Smat, DDIM, (long long)DDIM * DDIM, DDIM, DDIM, 256);
    p.alphaMul = -1.0f; p.alphaInvC = 1; p.cPtr = W.cPtr; p.addI = 1;
    launch_gemm_t<1, 0, 0>(st, p, 2);
  }
  // z0 = soft(y, thr)
  k_soft0<<<(int)CDIV(NBt * LP * DDIM, 256), 256, 0, st>>>(Y, THR, ZA, NBt * LP * DDIM);

  // 5 LISTA iterations: z = soft(z @ S + y/c, thr)
  const float* zin = ZA; float* zout = ZB;
  for (int t = 0; t < 5; ++t) {
    GemmP p = mkgemm(zin, DDIM, LP * DDIM, Smat, DDIM, (long long)DDIM * DDIM,
                     zout, DDIM, LP * DDIM, (int)LP, DDIM, DDIM);
    p.cPtr = W.cPtr;
    p.E1 = Y;   p.ldE1 = DDIM; p.sE1 = LP * DDIM;
    p.E2 = THR; p.ldE2 = DDIM; p.sE2 = LP * DDIM;
    launch_gemm_t<0, 0, 1>(st, p, 2);
    const float* tswap = zin; zin = zout; zout = (float*)tswap;
  }
  // x_pred = clip(z @ Dcat^T, 0, 1) * Wg ; B = Dcat stored [N=256? no: [256,624]] -> transB
  {
    GemmP p = mkgemm(zin, DDIM, LP * DDIM, DCAT, DDIM, 256LL * DDIM,
                     XP, 256, LP * 256, (int)LP, 256, DDIM);
    p.E1 = WG; p.ldE1 = 256; p.sE1 = LP * 256;
    launch_gemm_t<0, 1, 2>(st, p, 2);
  }
  // overlap-add fold + normalize
  k_fold_div<<<(int)CDIV(NBt * 16384, 256), 256, 0, st>>>(XP, WG, outimg);
}

// ---------------------------------------------------------------------------
// Entry point
// ---------------------------------------------------------------------------
extern "C" void kernel_launch(void* const* d_in, const int* in_sizes, int n_in,
                              void* d_out, int out_size, void* d_ws, size_t ws_size,
                              hipStream_t stream) {
  (void)in_sizes; (void)n_in; (void)out_size; (void)ws_size;
  const float* in[44];
  for (int i = 0; i < 44; ++i) in[i] = (const float*)d_in[i];

  const float* X    = in[0];
  const float* Dict = in[1];
  const float* cP   = in[2];
  float* ws  = (float*)d_ws;
  float* out = (float*)d_out;
  float* RES = ws + RES_OFF;

  StageW W1{}, W2{};
  // shared sdict MLP + CBAM + gating weights
  W1.s1w = in[3];  W1.s1b = in[4];  W1.s2w = in[5];  W1.s2b = in[6];
  W1.s3w = in[7];  W1.s3b = in[8];  W1.s4w = in[9];  W1.s4b = in[10];
  W1.caw1 = in[41]; W1.caw2 = in[42]; W1.saw = in[43];
  W1.w1w = in[35]; W1.w1b = in[36]; W1.w2w = in[37]; W1.w2b = in[38];
  W1.w3w = in[39]; W1.w3b = in[40];
  W1.Dict = Dict; W1.cPtr = cP;
  W2 = W1;
  // stage 1: lam = a*, pd = p*
  W1.l1w = in[11]; W1.l1b = in[12]; W1.l2w = in[13]; W1.l2b = in[14];
  W1.l3w = in[15]; W1.l3b = in[16];
  W1.p1w = in[17]; W1.p1b = in[18]; W1.p2w = in[19]; W1.p2b = in[20];
  W1.p3w = in[21]; W1.p3b = in[22];
  // stage 2: lam = b*, pd = q*
  W2.l1w = in[23]; W2.l1b = in[24]; W2.l2w = in[25]; W2.l2b = in[26];
  W2.l3w = in[27]; W2.l3b = in[28];
  W2.p1w = in[29]; W2.p1b = in[30]; W2.p2w = in[31]; W2.p2b = in[32];
  W2.p3w = in[33]; W2.p3b = in[34];

  run_stage(stream, X,   RES, ws, W1);
  run_stage(stream, RES, out, ws, W2);
}
